// JetBlock_51651276701956
// MI455X (gfx1250) — matrix operand order repaired and checked by reading, not compiled, not run on hardware
//
#include <hip/hip_runtime.h>
#include <hip/hip_bf16.h>

// Problem constants (match reference)
#define B_   2
#define T_   4096
#define DM_  2048      // generator input width (K of GEMM1)
#define D_   4096      // conv channels
#define W_   4         // taps
#define DG_  512       // generator hidden (N of GEMM1, K of GEMM2)
#define MTOT (B_*T_)   // 8192 tokens
#define NK_  (D_*W_)   // 16384 (N of GEMM2)

typedef __bf16 bf16;
typedef __attribute__((ext_vector_type(16))) __bf16 bf16x16;
typedef __attribute__((ext_vector_type(8)))  __bf16 bf16x8;
typedef __attribute__((ext_vector_type(4)))  __bf16 bf16x4;
typedef __attribute__((ext_vector_type(8)))  float  f32x8;
typedef __attribute__((ext_vector_type(4)))  float  f32x4;
typedef __attribute__((ext_vector_type(4)))  int    i32x4;

// ---- CDNA5 async global->LDS copy (ASYNCcnt-tracked), with safe fallback ----
#if __has_builtin(__builtin_amdgcn_global_load_async_to_lds_b128)
#define HAS_ASYNC_LDS 1
#else
#define HAS_ASYNC_LDS 0
#endif

__device__ __forceinline__ void async_copy_b128(const void* gsrc, void* ldst) {
#if HAS_ASYNC_LDS
    // Builtin signature (from probe diagnostics): param 1 is a non-const
    // global-AS int4*, so: (global int4* src, local int4* dst, offset, cpol).
    __builtin_amdgcn_global_load_async_to_lds_b128(
        (__attribute__((address_space(1))) i32x4*)gsrc,
        (__attribute__((address_space(3))) i32x4*)ldst,
        /*offset=*/0, /*cpol=*/0);
#else
    *(bf16x8*)ldst = *(const bf16x8*)gsrc;
#endif
}

__device__ __forceinline__ void wait_async0() {
#if HAS_ASYNC_LDS
#if __has_builtin(__builtin_amdgcn_s_wait_asynccnt)
    __builtin_amdgcn_s_wait_asynccnt(0);
#else
    asm volatile("s_wait_asynccnt 0x0" ::: "memory");
#endif
#endif
}

__device__ __forceinline__ bf16x4 cvt_f4_bf4(f32x4 v) {
    bf16x4 r;
    r[0] = (bf16)v.x; r[1] = (bf16)v.y; r[2] = (bf16)v.z; r[3] = (bf16)v.w;
    return r;
}

__device__ __forceinline__ bf16x16 frag_cat(bf16x8 lo, bf16x8 hi) {
    bf16x16 a;
#pragma unroll
    for (int e = 0; e < 8; ++e) { a[e] = lo[e]; a[e + 8] = hi[e]; }
    return a;
}

// Fast SiLU: v_exp_f32 + v_add + v_rcp_f32 + v_mul.
__device__ __forceinline__ float silu(float v) {
    return v * __builtin_amdgcn_rcpf(1.0f + __expf(-v));
}

// ---------------------------------------------------------------------------
// Kernel 0: one-shot w2 f32 -> bf16 (removes per-M-tile re-conversion from the
// GEMM2 hot loop; 34 MB read + 17 MB write ~ 2 us at 23.3 TB/s).
// ---------------------------------------------------------------------------
__global__ void jet_cvt_w2(const float* __restrict__ w2, bf16* __restrict__ w2b)
{
    size_t i = ((size_t)blockIdx.x * 256 + threadIdx.x) * 4;
    f32x4 v = *(const f32x4*)&w2[i];
    *(bf16x4*)&w2b[i] = cvt_f4_bf4(v);
}

// ---------------------------------------------------------------------------
// Kernel 1: h = silu(gen @ w1^T), M=8192 K=2048 N=512, output bf16.
// 256 threads (8 waves), tile 128(M) x 64(N), K-step 32, double-buffered LDS.
// Wave grid 4(M) x 2(N); each wave computes 2x2 WMMA 16x16 tiles (32x32).
// ---------------------------------------------------------------------------
__launch_bounds__(256, 2)
__global__ void jet_gemm1_silu(const float* __restrict__ gen,
                               const float* __restrict__ w1,
                               bf16* __restrict__ h)
{
    constexpr int BM = 128, BN = 64, KT = 32;
    constexpr int OP = 72;  // padded output-repack row stride (bf16), 16B-aligned
    __shared__ __align__(16) bf16 Asl[2][BM * KT];  // 2 x 8 KB, [row][k]
    __shared__ __align__(16) bf16 Bsl[2][BN * KT];  // 2 x 4 KB, [n][k]
    __shared__ __align__(16) bf16 Osl[BM * OP];     // 18 KB epilogue repack

    const int tid  = threadIdx.x;
    const int lane = tid & 31;
    const int wave = tid >> 5;
    const int wm   = wave >> 1;          // 0..3
    const int wn   = wave & 1;           // 0..1
    const int m0   = blockIdx.y * BM;
    const int n0   = blockIdx.x * BN;
    const int l15  = lane & 15;
    const int lh   = lane >> 4;          // 0 or 1

    auto stage = [&](int k0, int buf) {
        // A: 128x32 f32 -> bf16 (1024 float4 units, 4 per thread)
#pragma unroll
        for (int i = 0; i < 4; ++i) {
            int u   = tid + i * 256;
            int row = u >> 3, c4 = (u & 7) * 4;
            f32x4 v = *(const f32x4*)&gen[(size_t)(m0 + row) * DM_ + k0 + c4];
            *(bf16x4*)&Asl[buf][row * KT + c4] = cvt_f4_bf4(v);
        }
        // B: 64x32 f32 -> bf16 (512 units, 2 per thread)
#pragma unroll
        for (int i = 0; i < 2; ++i) {
            int u   = tid + i * 256;
            int row = u >> 3, c4 = (u & 7) * 4;
            f32x4 v = *(const f32x4*)&w1[(size_t)(n0 + row) * DM_ + k0 + c4];
            *(bf16x4*)&Bsl[buf][row * KT + c4] = cvt_f4_bf4(v);
        }
    };

    f32x8 acc[2][2] = {};

    stage(0, 0);
    __syncthreads();

    constexpr int NKT = DM_ / KT;        // 64
    for (int kt = 0; kt < NKT; ++kt) {
        const int cur = kt & 1;
        if (kt + 1 < NKT) stage((kt + 1) * KT, cur ^ 1);  // overlap with WMMA

#pragma unroll
        for (int i = 0; i < 2; ++i) {
            // A frag: lane l<16 row=l K{0..7,16..23}; l>=16 K{8..15,24..31}
            int ar = wm * 32 + i * 16 + l15;
            bf16x16 a = frag_cat(
                *(const bf16x8*)&Asl[cur][ar * KT + lh * 8],
                *(const bf16x8*)&Asl[cur][ar * KT + 16 + lh * 8]);
#pragma unroll
            for (int j = 0; j < 2; ++j) {
                // B frag: column n=l15, K{0..7,16..23} (lh=0) / K{8..15,24..31} (lh=1)
                int br = wn * 32 + j * 16 + l15;
                bf16x16 b = frag_cat(
                    *(const bf16x8*)&Bsl[cur][br * KT + lh * 8],
                    *(const bf16x8*)&Bsl[cur][br * KT + 16 + lh * 8]);
                acc[i][j] = __builtin_amdgcn_wmma_f32_16x16x32_bf16(
                    false, a, false, b, (short)0, acc[i][j], false, false);
            }
        }
        __syncthreads();  // protects cur (readers done) and nxt (writes visible)
    }

    // Epilogue: SiLU -> bf16 -> LDS repack -> coalesced b128 stores.
#pragma unroll
    for (int i = 0; i < 2; ++i) {
#pragma unroll
        for (int j = 0; j < 2; ++j) {
            int nn = wn * 32 + j * 16 + l15;
#pragma unroll
            for (int r = 0; r < 8; ++r) {
                int mm = wm * 32 + i * 16 + lh * 8 + r;
                Osl[mm * OP + nn] = (bf16)silu(acc[i][j][r]);
            }
        }
    }
    __syncthreads();
#pragma unroll
    for (int i = 0; i < 4; ++i) {
        int u   = tid + i * 256;          // 1024 chunks of 8 bf16
        int row = u >> 3, c8 = (u & 7) * 8;
        bf16x8 vv = *(const bf16x8*)&Osl[row * OP + c8];
        *(bf16x8*)&h[(size_t)(m0 + row) * DG_ + n0 + c8] = vv;
    }
}

// ---------------------------------------------------------------------------
// Kernel 2: fused kflat = h @ w2b^T + b2 -> causal depthwise conv -> silu.
// M=8192 K=512 N=16384. Block tile: 32(M tokens) x 256(N = 64 ch x 4 taps).
// 8 waves: 2(M) x 4(N); each wave: 1x4 WMMA tiles (16x64). Double-buffered.
// A/B tiles are pure bf16 copies -> GLOBAL_LOAD_ASYNC_TO_LDS_B128 when
// available (ASYNCcnt-tracked, no VGPR round trip).
// kflat tile kept entirely in LDS (never hits HBM: saves >1 GB of traffic).
// ---------------------------------------------------------------------------
__launch_bounds__(256, 2)
__global__ void jet_gemm2_conv_silu(const bf16*  __restrict__ h,
                                    const bf16*  __restrict__ w2b,
                                    const float* __restrict__ b2,
                                    const float* __restrict__ x,
                                    float* __restrict__ out)
{
    constexpr int BM = 32, BN = 256, KT = 32;
    constexpr int CH = BN / W_;                 // 64 channels per block
    __shared__ __align__(16) bf16  Asl[2][BM * KT];         // 2 x 2 KB
    __shared__ __align__(16) bf16  Bsl[2][BN * KT];         // 2 x 16 KB
    __shared__ __align__(16) float Kfl[BM * BN];            // 32 KB kflat tile
    __shared__ __align__(16) float Xsl[(BM + W_ - 1) * CH]; // 8.75 KB x window

    const int tid  = threadIdx.x;
    const int lane = tid & 31;
    const int wave = tid >> 5;
    const int wm   = wave & 1;           // 0..1 (M)
    const int wn   = wave >> 1;          // 0..3 (N)
    const int m0   = blockIdx.y * BM;    // token tile (never crosses batch: T%BM==0)
    const int n0   = blockIdx.x * BN;
    const int d0   = n0 / W_;            // first channel of this block
    const int bb   = m0 / T_;
    const int t0   = m0 % T_;
    const int l15  = lane & 15;
    const int lh   = lane >> 4;

    // Stage x window early: rows r=0..34 map to token t0+r-3 (zero-pad at batch start).
    for (int u = tid; u < (BM + W_ - 1) * CH; u += 256) {
        int r = u / CH, c = u % CH;
        int t = t0 + r - (W_ - 1);
        float v = 0.0f;
        if (t >= 0) v = x[((size_t)bb * T_ + t) * D_ + d0 + c];
        Xsl[u] = v;
    }

    // Prefetch bias (4 scalar loads/lane, latency hidden behind GEMM).
    float bias[4];
#pragma unroll
    for (int j = 0; j < 4; ++j) bias[j] = b2[n0 + wn * 64 + j * 16 + l15];

    auto stage = [&](int k0, int buf) {
        // A: 32x32 bf16 from h (128 x 16B chunks, threads 0..127)
        if (tid < 128) {
            int row = tid >> 2, c8 = (tid & 3) * 8;
            async_copy_b128(&h[(size_t)(m0 + row) * DG_ + k0 + c8],
                            &Asl[buf][row * KT + c8]);
        }
        // B: 256x32 bf16 from w2b (1024 x 16B chunks, 4 per thread)
#pragma unroll
        for (int i = 0; i < 4; ++i) {
            int u   = tid + i * 256;
            int row = u >> 2, c8 = (u & 3) * 8;
            async_copy_b128(&w2b[(size_t)(n0 + row) * DG_ + k0 + c8],
                            &Bsl[buf][row * KT + c8]);
        }
    };

    f32x8 acc[4] = {};

    stage(0, 0);
    wait_async0();
    __syncthreads();

    constexpr int NKT = DG_ / KT;        // 16
    for (int kt = 0; kt < NKT; ++kt) {
        const int cur = kt & 1;
        if (kt + 1 < NKT) stage((kt + 1) * KT, cur ^ 1);

        int ar = wm * 16 + l15;
        bf16x16 a = frag_cat(
            *(const bf16x8*)&Asl[cur][ar * KT + lh * 8],
            *(const bf16x8*)&Asl[cur][ar * KT + 16 + lh * 8]);
#pragma unroll
        for (int j = 0; j < 4; ++j) {
            int br = wn * 64 + j * 16 + l15;
            bf16x16 b = frag_cat(
                *(const bf16x8*)&Bsl[cur][br * KT + lh * 8],
                *(const bf16x8*)&Bsl[cur][br * KT + 16 + lh * 8]);
            acc[j] = __builtin_amdgcn_wmma_f32_16x16x32_bf16(
                false, a, false, b, (short)0, acc[j], false, false);
        }
        wait_async0();     // next buffer's async fills complete (ASYNCcnt)
        __syncthreads();   // publish to all waves / retire readers of cur
    }

    // Spill kflat tile (+bias) to LDS
#pragma unroll
    for (int j = 0; j < 4; ++j) {
        int nn = wn * 64 + j * 16 + l15;
#pragma unroll
        for (int r = 0; r < 8; ++r) {
            int mm = wm * 16 + lh * 8 + r;
            Kfl[mm * BN + nn] = acc[j][r] + bias[j];
        }
    }
    __syncthreads();

    // Depthwise conv + SiLU: 32x64 outputs, 8 per thread.
    // kflat[t][c*4 + w] is one aligned float4 per output (ds_load_b128).
#pragma unroll
    for (int i = 0; i < 8; ++i) {
        int u  = tid + i * 256;
        int tl = u >> 6;          // token within tile (CH == 64)
        int c  = u & (CH - 1);    // channel within tile
        f32x4 kf = *(const f32x4*)&Kfl[tl * BN + c * W_];
        float y = kf.x * Xsl[(tl + 0) * CH + c]
                + kf.y * Xsl[(tl + 1) * CH + c]
                + kf.z * Xsl[(tl + 2) * CH + c]
                + kf.w * Xsl[(tl + 3) * CH + c];
        out[((size_t)(m0 + tl)) * D_ + d0 + c] = silu(y);
    }
}

extern "C" void kernel_launch(void* const* d_in, const int* in_sizes, int n_in,
                              void* d_out, int out_size, void* d_ws, size_t ws_size,
                              hipStream_t stream) {
    const float* x   = (const float*)d_in[0];   // [B,T,D]
    const float* gen = (const float*)d_in[1];   // [B,T,DM]
    const float* w1  = (const float*)d_in[2];   // [DG,DM]
    const float* w2  = (const float*)d_in[3];   // [D*W,DG]
    const float* b2  = (const float*)d_in[4];   // [D*W]
    float* out = (float*)d_out;                 // [B,T,D] f32

    // Workspace layout: h bf16 [8192,512] (8 MB), w2b bf16 [16384,512] (16.8 MB)
    bf16* hws  = (bf16*)d_ws;
    bf16* w2b  = (bf16*)((char*)d_ws + (size_t)MTOT * DG_ * sizeof(bf16));

    jet_cvt_w2<<<(NK_ * DG_) / (256 * 4), 256, 0, stream>>>(w2, w2b);

    dim3 g1(DG_ / 64, MTOT / 128);              // (8, 64)
    jet_gemm1_silu<<<g1, 256, 0, stream>>>(gen, w1, hws);

    dim3 g2(NK_ / 256, MTOT / 32);              // (64, 256)
    jet_gemm2_conv_silu<<<g2, 256, 0, stream>>>(hws, w2b, b2, x, out);
}
